// OrientationPenalty_82240033784130
// MI455X (gfx1250) — matrix operand order repaired and checked by reading, not compiled, not run on hardware
//
#include <hip/hip_runtime.h>
#include <math.h>

#define D        512
#define NROWS    65536
#define NF       65536.0f
#define NM1      65535.0f
#define EPSF     1e-8f
#define TWO_PI_E 17.07946844534713f   // 2*pi*e

typedef __attribute__((ext_vector_type(16))) __bf16 v16bf;
typedef __attribute__((ext_vector_type(8)))  float  v8f;
typedef __attribute__((ext_vector_type(2)))  float  v2f;

// ---------------- workspace layout (float offsets) ----------------
#define OFF_G    0          // 512*512 : Gram X^T X  -> centered M (in place)
#define OFF_S    262144     // 512     : column sums of X
#define OFF_ACC  262656     // 16      : [0]=sum(A) [1]=L_entropy_num [2]=L_residual_num
#define OFF_ENT  262672     // 512
#define OFF_XSTD 263184     // 512
#define OFF_RSTD 263696     // 512
#define OFF_P    264208     // 512*512 : P = M @ A
#define ZERO_N   262672     // zero G + S + ACC region

// async transposing copy: LDS[loff] = *gp   (per-lane dest, ASYNCcnt-tracked)
__device__ __forceinline__ void async_copy_b32(unsigned loff, const float* gp) {
  asm volatile("global_load_async_to_lds_b32 %0, %1, off"
               :: "v"(loff), "v"(gp) : "memory");
}
__device__ __forceinline__ void wait_async0() {
  asm volatile("s_wait_asynccnt 0x0" ::: "memory");
}

// -------------------------------------------------------------------
__global__ void k_zero(float* ws, int n) {
  int i = blockIdx.x * blockDim.x + threadIdx.x;
  if (i < n) ws[i] = 0.0f;
}

// column sums: block b covers 64 rows, thread j = column (coalesced)
__global__ void k_colsum(const float* __restrict__ X, float* __restrict__ s) {
  int j  = threadIdx.x;            // 512 threads
  int r0 = blockIdx.x * 64;
  float acc = 0.0f;
  for (int r = 0; r < 64; ++r) acc += X[(size_t)(r0 + r) * D + j];
  atomicAdd(&s[j], acc);
}

// -------------------------------------------------------------------
// Gram: G += X^T X over a K-chunk. grid (8,8,16), block 128 (4 waves).
// K-slabs of 32 rows stream global->LDS via transposing async copies
// (double buffered, rolling-pointer addressing); fragments are read
// K-contiguous from LDS and converted to bf16 at the WMMA operands.
__global__ void __launch_bounds__(128)
k_gram(const float* __restrict__ X, float* __restrict__ G, int kChunk) {
  // Tf[buf][panel][col 0..63][k 0..31 (+pad to 36)]  f32, 36 KB total
  __shared__ float Tf[2][2][64][36];
  constexpr unsigned BUF_STRIDE   = 2u * 64u * 36u * 4u;  // 18432 B
  constexpr unsigned PANEL_STRIDE = 64u * 36u * 4u;       //  9216 B

  const int lane = threadIdx.x & 31;
  const int wave = threadIdx.x >> 5;
  const int qi = wave >> 1, qj = wave & 1;
  const int m  = lane & 15;
  const int hh = lane >> 4;
  const int I0b = blockIdx.x * 64;
  const int J0b = blockIdx.y * 64;
  const int k0 = blockIdx.z * kChunk;
  const int k1 = k0 + kChunk;

  const int t     = threadIdx.x;
  const int ccol  = t & 63;        // fixed column handled by this thread
  const int kbase = t >> 6;        // 0 or 1 (even/odd row start)

  // hoisted per-thread bases
  const unsigned ldsBase = (unsigned)(size_t)&Tf[0][0][ccol][kbase];
  const float* g0 = X + (size_t)(k0 + kbase) * D + I0b + ccol;  // next slab, panel 0
  const float* g1 = X + (size_t)(k0 + kbase) * D + J0b + ccol;  // next slab, panel 1

  v8f zero = {};
  v8f acc[2][2];
  for (int r = 0; r < 2; ++r)
    for (int c = 0; c < 2; ++c) acc[r][c] = zero;

  // one panel = 16 copies, rows kbase, kbase+2, ...: +8 B in LDS, +2 rows global
  auto issue_panel = [&](unsigned lbase, const float* p) {
#pragma unroll
    for (int i = 0; i < 16; ++i) {
      async_copy_b32(lbase + 8u * (unsigned)i, p);
      p += 2 * D;
    }
  };

  // prologue: slab 0 into buffer 0
  issue_panel(ldsBase, g0);
  issue_panel(ldsBase + PANEL_STRIDE, g1);
  g0 += 32 * D;
  g1 += 32 * D;

  int cur = 0;
  for (int kb = k0; kb < k1; kb += 32, cur ^= 1) {
    wait_async0();                         // slab `cur` landed in LDS
    __syncthreads();                       // visible to all waves; prev reads done
    if (kb + 32 < k1) {                    // stream next slab into other buffer
      unsigned l = ldsBase + (cur ? 0u : BUF_STRIDE);
      issue_panel(l, g0);
      issue_panel(l + PANEL_STRIDE, g1);
      g0 += 32 * D;
      g1 += 32 * D;
    }

    // A-frag (16x32): A[m,k] = X[k, I0 + r*16 + m]; elems 0..7 -> K=hh*8+e,
    // elems 8..15 -> K=16+hh*8+e. K-contiguous in Tf -> b128-mergeable.
    v16bf a[2], b[2];
#pragma unroll
    for (int r = 0; r < 2; ++r) {
      const float* lo = &Tf[cur][0][qi * 32 + r * 16 + m][hh * 8];
      const float* hi = lo + 16;
#pragma unroll
      for (int e = 0; e < 8; ++e) {
        a[r][e]     = (__bf16)lo[e];
        a[r][8 + e] = (__bf16)hi[e];
      }
    }
  // B-frag (32x16): B[k,n] = X[k, J0 + c*16 + n]; same K order as A-frag.
#pragma unroll
    for (int c = 0; c < 2; ++c) {
      const float* p = &Tf[cur][1][qj * 32 + c * 16 + m][hh * 8];
#pragma unroll
      for (int e = 0; e < 16; ++e)
        b[c][e] = (__bf16)p[(e < 8) ? e : e + 8];
    }
#pragma unroll
    for (int r = 0; r < 2; ++r)
#pragma unroll
      for (int c = 0; c < 2; ++c)
        acc[r][c] = __builtin_amdgcn_wmma_f32_16x16x32_bf16(
            false, a[r], false, b[c], (short)0, acc[r][c], false, false);
  }

  // split-K accumulate: C/D layout -> VGPR rr holds (M = rr + 8*hh, N = m)
#pragma unroll
  for (int r = 0; r < 2; ++r)
#pragma unroll
    for (int c = 0; c < 2; ++c) {
      int col = J0b + qj * 32 + c * 16 + m;
#pragma unroll
      for (int rr = 0; rr < 8; ++rr) {
        int row = I0b + qi * 32 + r * 16 + hh * 8 + rr;
        atomicAdd(&G[(size_t)row * D + col], acc[r][c][rr]);
      }
    }
}

// M = G - s s^T / N (in place)
__global__ void k_center(float* __restrict__ G, const float* __restrict__ s) {
  int i = blockIdx.x, j = threadIdx.x;
  G[(size_t)i * D + j] -= s[i] * s[j] / NF;
}

// entropy + X_std from diag(M)
__global__ void k_stats(const float* __restrict__ Mm, float* __restrict__ ent,
                        float* __restrict__ xstd) {
  int i = threadIdx.x;  // 512
  float v = Mm[(size_t)i * D + i] / NM1;
  ent[i]  = 0.5f * logf(TWO_PI_E * (v + EPSF));
  xstd[i] = sqrtf(fmaxf(v, 0.0f)) + EPSF;
}

// -------------------------------------------------------------------
// P = M @ A, full f32 precision via v_wmma_f32_16x16x4_f32.
__global__ void __launch_bounds__(128)
k_smallgemm(const float* __restrict__ Mm, const float* __restrict__ A,
            float* __restrict__ P) {
  const int lane = threadIdx.x & 31;
  const int wave = threadIdx.x >> 5;
  const int qi = wave >> 1, qj = wave & 1;
  const int m  = lane & 15;
  const int hh = lane >> 4;
  const int I0 = blockIdx.x * 64 + qi * 32;
  const int J0 = blockIdx.y * 64 + qj * 32;

  v8f zero = {};
  v8f acc[2][2];
  for (int r = 0; r < 2; ++r)
    for (int c = 0; c < 2; ++c) acc[r][c] = zero;

  for (int kb = 0; kb < D; kb += 4) {
    v2f a[2], b[2];
#pragma unroll
    for (int r = 0; r < 2; ++r) {
      const float* rp = Mm + (size_t)(I0 + r * 16 + m) * D + kb + 2 * hh;
      a[r][0] = rp[0];
      a[r][1] = rp[1];
    }
#pragma unroll
    for (int c = 0; c < 2; ++c) {
      const float* cp = A + (size_t)(kb + 2 * hh) * D + (J0 + c * 16 + m);
      b[c][0] = cp[0];
      b[c][1] = cp[D];
    }
#pragma unroll
    for (int r = 0; r < 2; ++r)
#pragma unroll
      for (int c = 0; c < 2; ++c)
        acc[r][c] = __builtin_amdgcn_wmma_f32_16x16x4_f32(
            false, a[r], false, b[c], (short)0, acc[r][c], false, false);
  }

#pragma unroll
  for (int r = 0; r < 2; ++r)
#pragma unroll
    for (int c = 0; c < 2; ++c) {
      int col = J0 + c * 16 + m;
#pragma unroll
      for (int rr = 0; rr < 8; ++rr) {
        int row = I0 + r * 16 + hh * 8 + rr;
        P[(size_t)row * D + col] = acc[r][c][rr];
      }
    }
}

// res_std[j] from B = M - P:  res_var[j] = (B_jj - sum_i A_ij * B_ij) / (N-1)
__global__ void k_resvar(const float* __restrict__ Mm, const float* __restrict__ Pm,
                         const float* __restrict__ A, float* __restrict__ rstd) {
  __shared__ float red[256];
  int j = blockIdx.x;
  float acc = 0.0f;
  for (int i = threadIdx.x; i < D; i += 256) {
    float bij = Mm[(size_t)i * D + j] - Pm[(size_t)i * D + j];
    acc += A[(size_t)i * D + j] * bij;
  }
  red[threadIdx.x] = acc;
  __syncthreads();
  for (int s2 = 128; s2 > 0; s2 >>= 1) {
    if (threadIdx.x < s2) red[threadIdx.x] += red[threadIdx.x + s2];
    __syncthreads();
  }
  if (threadIdx.x == 0) {
    float bjj = Mm[(size_t)j * D + j] - Pm[(size_t)j * D + j];
    float rv  = (bjj - red[0]) / NM1;
    rstd[j]   = sqrtf(fmaxf(rv, 0.0f)) + EPSF;
  }
}

// weighted sums: sum(A), sum A*relu(e_i-e_j), sum_{i!=j} A*|cov/(xs_i*rs_j)|
__global__ void k_losses(const float* __restrict__ A, const float* __restrict__ Mm,
                         const float* __restrict__ Pm, const float* __restrict__ ent,
                         const float* __restrict__ xstd, const float* __restrict__ rstd,
                         float* __restrict__ acc) {
  __shared__ float rA[512], rE[512], rR[512];
  int i = blockIdx.x, j = threadIdx.x;
  float a  = A[(size_t)i * D + j];
  float le = a * fmaxf(ent[i] - ent[j], 0.0f);
  float lr = 0.0f;
  if (i != j) {
    float cov = (Mm[(size_t)i * D + j] - Pm[(size_t)i * D + j]) / NF;
    lr = a * fabsf(cov / (xstd[i] * rstd[j]));
  }
  rA[j] = a; rE[j] = le; rR[j] = lr;
  __syncthreads();
  for (int s2 = 256; s2 > 0; s2 >>= 1) {
    if (j < s2) { rA[j] += rA[j + s2]; rE[j] += rE[j + s2]; rR[j] += rR[j + s2]; }
    __syncthreads();
  }
  if (j == 0) {
    atomicAdd(&acc[0], rA[0]);
    atomicAdd(&acc[1], rE[0]);
    atomicAdd(&acc[2], rR[0]);
  }
}

__global__ void k_out(const float* __restrict__ acc, float* __restrict__ out) {
  float denom = acc[0] + EPSF;
  out[0] = 0.1f * (acc[1] / denom + 0.5f * (acc[2] / denom));
}

// -------------------------------------------------------------------
extern "C" void kernel_launch(void* const* d_in, const int* in_sizes, int n_in,
                              void* d_out, int out_size, void* d_ws, size_t ws_size,
                              hipStream_t stream) {
  (void)in_sizes; (void)n_in; (void)out_size; (void)ws_size;
  const float* X = (const float*)d_in[0];
  const float* A = (const float*)d_in[1];
  float* out = (float*)d_out;
  float* ws  = (float*)d_ws;

  float* G   = ws + OFF_G;
  float* s   = ws + OFF_S;
  float* acc = ws + OFF_ACC;
  float* ent = ws + OFF_ENT;
  float* xs  = ws + OFF_XSTD;
  float* rs  = ws + OFF_RSTD;
  float* P   = ws + OFF_P;

  k_zero<<<(ZERO_N + 511) / 512, 512, 0, stream>>>(ws, ZERO_N);
  k_colsum<<<NROWS / 64, 512, 0, stream>>>(X, s);

  dim3 gg(8, 8, 16);
  k_gram<<<gg, 128, 0, stream>>>(X, G, NROWS / 16);

  k_center<<<512, 512, 0, stream>>>(G, s);
  k_stats<<<1, 512, 0, stream>>>(G, ent, xs);

  dim3 gs(8, 8);
  k_smallgemm<<<gs, 128, 0, stream>>>(G, A, P);

  k_resvar<<<512, 256, 0, stream>>>(G, P, A, rs);
  k_losses<<<512, 512, 0, stream>>>(A, G, P, ent, xs, rs, acc);
  k_out<<<1, 1, 0, stream>>>(acc, out);
}
